// QuantizedLinearTriton_52982716564218
// MI455X (gfx1250) — matrix operand order repaired and checked by reading, not compiled, not run on hardware
//
#include <hip/hip_runtime.h>
#include <hip/hip_bf16.h>
#include <stdint.h>

// ---------------------------------------------------------------------------
// QuantizedLinear: out[b,s,o] = x[b,s,k] * (centroids[nibble(Wp)] * scale).T
// M = 4*2048 = 8192, N = 4096, K = 4096, scale block = 64 along K.
// Strategy: (0) x f32 -> bf16 in ws, (1) dequant packed 4-bit W -> bf16 in ws,
// (2) tiled bf16 WMMA GEMM with ASYNC global->LDS staging (ASYNCcnt) and
//     v_wmma_f32_16x16x32_bf16 compute, f32 accumulate.
// ---------------------------------------------------------------------------

typedef __bf16 bf16_t;
typedef __bf16 v16bf __attribute__((ext_vector_type(16)));
typedef __bf16 v8bf  __attribute__((ext_vector_type(8)));
typedef float  v8f   __attribute__((ext_vector_type(8)));
typedef float  v4f   __attribute__((ext_vector_type(4)));

typedef __attribute__((address_space(3))) void lds_void;

#define M_DIM 8192
#define N_DIM 4096
#define K_DIM 4096
#define SCALE_BLK 64

#define TM 128
#define TN 128
#define TK 32
#define NK (K_DIM / TK)      // 128 K-steps
#define LDP 40               // LDS row pitch in halfs (80 B, multiple of 16 B)

// --- gfx1250 async global->LDS copy (ASYNCcnt), inline-asm form ------------
// global_load_async_to_lds_b128 vdst(=LDS byte addr), v[addr:addr+1], off
__device__ __forceinline__ void async_copy_b128(void* lds, const void* g) {
  uint32_t lds_off = (uint32_t)(__SIZE_TYPE__)(lds_void*)lds;
  uint64_t gaddr   = (uint64_t)(__SIZE_TYPE__)g;
  asm volatile("global_load_async_to_lds_b128 %0, %1, off"
               :: "v"(lds_off), "v"(gaddr)
               : "memory");
}

__device__ __forceinline__ void wait_async0() {
#if __has_builtin(__builtin_amdgcn_s_wait_asynccnt)
  __builtin_amdgcn_s_wait_asynccnt(0);
#else
  asm volatile("s_wait_asynccnt 0x0" ::: "memory");
#endif
}

// ---------------------------------------------------------------------------
// Phase 0: x f32 -> bf16 (8 elements / thread, b128 stores)
// ---------------------------------------------------------------------------
__global__ __launch_bounds__(256) void x_to_bf16(const float* __restrict__ x,
                                                 bf16_t* __restrict__ xo) {
  size_t i = ((size_t)blockIdx.x * 256 + threadIdx.x) * 8;
  v4f a = *(const v4f*)(x + i);
  v4f b = *(const v4f*)(x + i + 4);
  v8bf o;
  o[0] = (bf16_t)a[0]; o[1] = (bf16_t)a[1]; o[2] = (bf16_t)a[2]; o[3] = (bf16_t)a[3];
  o[4] = (bf16_t)b[0]; o[5] = (bf16_t)b[1]; o[6] = (bf16_t)b[2]; o[7] = (bf16_t)b[3];
  *(v8bf*)(xo + i) = o;
}

// ---------------------------------------------------------------------------
// Phase 1: dequant W. Each thread: one u32 of packed data = 4 bytes = 8 weights.
// byte j of row covers K=2j (HIGH nibble first) and K=2j+1 (low nibble).
// ---------------------------------------------------------------------------
__global__ __launch_bounds__(256) void dequant_w(const uint8_t* __restrict__ wp,
                                                 const float* __restrict__ cent,
                                                 const float* __restrict__ scales,
                                                 bf16_t* __restrict__ wout) {
  __shared__ float lut[16];
  int tid = threadIdx.x;
  if (tid < 16) lut[tid] = cent[tid];
  __syncthreads();

  size_t gid = (size_t)blockIdx.x * 256 + tid;   // chunk id, K/8 = 512 chunks/row
  int row = (int)(gid >> 9);
  int k0  = ((int)gid & 511) * 8;
  float s = scales[(size_t)row * (K_DIM / SCALE_BLK) + (k0 >> 6)];
  uint32_t packed = *(const uint32_t*)(wp + (size_t)row * (K_DIM / 2) + (k0 >> 1));

  v8bf o;
#pragma unroll
  for (int b = 0; b < 4; ++b) {
    uint32_t byte = (packed >> (8 * b)) & 0xFFu;
    float hi = lut[byte >> 4] * s;     // first element of pair
    float lo = lut[byte & 0xF] * s;    // second element
    o[2 * b]     = (bf16_t)hi;
    o[2 * b + 1] = (bf16_t)lo;
  }
  *(v8bf*)(wout + (size_t)row * K_DIM + k0) = o;
}

// ---------------------------------------------------------------------------
// Phase 2: bf16 WMMA GEMM. C[M][N] = A[M][K] * B[N][K]^T
// 256 threads (8 wave32), 128x128 tile, TK=32, double-buffered LDS,
// async global->LDS staging. Wave (wm 0..1, wn 0..3) owns 64x32 output
// (4x2 tiles of 16x16 -> 8 WMMA per K-step).
// ---------------------------------------------------------------------------
__global__ __launch_bounds__(256) void gemm_bf16(const bf16_t* __restrict__ A,
                                                 const bf16_t* __restrict__ B,
                                                 float* __restrict__ C) {
  __shared__ __attribute__((aligned(16))) bf16_t sA[2][TM][LDP];
  __shared__ __attribute__((aligned(16))) bf16_t sB[2][TN][LDP];

  const int tid  = threadIdx.x;
  const int lane = tid & 31;
  const int wave = tid >> 5;
  const int wm = wave & 1;   // M offset wm*64
  const int wn = wave >> 1;  // N offset wn*32
  const int m0 = blockIdx.y * TM;
  const int n0 = blockIdx.x * TN;

  // staging: thread -> (row = tid>>1, 32-byte half = tid&1); 2x async b128 each
  const int srow  = tid >> 1;
  const int shalf = (tid & 1) * 16;  // offset in halfs
  const bf16_t* aRow = A + (size_t)(m0 + srow) * K_DIM + shalf;
  const bf16_t* bRow = B + (size_t)(n0 + srow) * K_DIM + shalf;

  const int lr   = lane & 15;
  const int aklo = (lane < 16) ? 0 : 8;    // A: lanes 0-15 hold K 0-7 & 16-23
  const int bklo = (lane < 16) ? 0 : 8;    // B: lanes 0-15 hold K 0-7 & 16-23

  v8f acc[4][2] = {};

  // prologue: async-stage tile 0 into buffer 0
  {
    async_copy_b128(&sA[0][srow][shalf],     aRow);
    async_copy_b128(&sA[0][srow][shalf + 8], aRow + 8);
    async_copy_b128(&sB[0][srow][shalf],     bRow);
    async_copy_b128(&sB[0][srow][shalf + 8], bRow + 8);
  }
  wait_async0();
  __syncthreads();

  int buf = 0;
  for (int kt = 0; kt < NK; ++kt) {
    // fire async DMA for the next tile into the other buffer; overlaps WMMA
    if (kt + 1 < NK) {
      const bf16_t* ap = aRow + (size_t)(kt + 1) * TK;
      const bf16_t* bp = bRow + (size_t)(kt + 1) * TK;
      int nb = buf ^ 1;
      async_copy_b128(&sA[nb][srow][shalf],     ap);
      async_copy_b128(&sA[nb][srow][shalf + 8], ap + 8);
      async_copy_b128(&sB[nb][srow][shalf],     bp);
      async_copy_b128(&sB[nb][srow][shalf + 8], bp + 8);
    }
    if (kt + 2 < NK) {
      __builtin_prefetch(aRow + (size_t)(kt + 2) * TK, 0, 1);  // global_prefetch_b8
      __builtin_prefetch(bRow + (size_t)(kt + 2) * TK, 0, 1);
    }

    // load fragments from LDS (ISA 7.12.2 layouts) and issue 8 WMMAs
    v16bf afr[4];
#pragma unroll
    for (int i = 0; i < 4; ++i) {
      const bf16_t* p = &sA[buf][wm * 64 + i * 16 + lr][0];
      v8bf lo = *(const v8bf*)(p + aklo);
      v8bf hi = *(const v8bf*)(p + aklo + 16);
      afr[i] = __builtin_shufflevector(lo, hi, 0, 1, 2, 3, 4, 5, 6, 7,
                                       8, 9, 10, 11, 12, 13, 14, 15);
    }
    v16bf bfr[2];
#pragma unroll
    for (int j = 0; j < 2; ++j) {
      const bf16_t* p = &sB[buf][wn * 32 + j * 16 + lr][0];
      v8bf lo = *(const v8bf*)(p + bklo);
      v8bf hi = *(const v8bf*)(p + bklo + 16);
      bfr[j] = __builtin_shufflevector(lo, hi, 0, 1, 2, 3, 4, 5, 6, 7,
                                       8, 9, 10, 11, 12, 13, 14, 15);
    }
#pragma unroll
    for (int i = 0; i < 4; ++i)
#pragma unroll
      for (int j = 0; j < 2; ++j)
        acc[i][j] = __builtin_amdgcn_wmma_f32_16x16x32_bf16(
            false, afr[i], false, bfr[j], (short)0, acc[i][j], false, false);

    // next tile's DMA must have landed before anyone reads it
    if (kt + 1 < NK) wait_async0();
    __syncthreads();
    buf ^= 1;
  }

  // epilogue: C/D layout = VGPR r: lanes0-15 -> M=r, lanes16-31 -> M=8+r
#pragma unroll
  for (int i = 0; i < 4; ++i) {
#pragma unroll
    for (int j = 0; j < 2; ++j) {
      int n  = n0 + wn * 32 + j * 16 + lr;
      int mb = m0 + wm * 64 + i * 16 + ((lane >> 4) << 3);
      float* cp = C + (size_t)mb * N_DIM + n;
#pragma unroll
      for (int r = 0; r < 8; ++r) cp[(size_t)r * N_DIM] = acc[i][j][r];
    }
  }
}

// ---------------------------------------------------------------------------
extern "C" void kernel_launch(void* const* d_in, const int* in_sizes, int n_in,
                              void* d_out, int out_size, void* d_ws, size_t ws_size,
                              hipStream_t stream) {
  const float*   x      = (const float*)d_in[0];
  const uint8_t* wp     = (const uint8_t*)d_in[1];  // packed 4-bit, uint8
  const float*   cent   = (const float*)d_in[2];
  const float*   scales = (const float*)d_in[3];
  float*         out    = (float*)d_out;

  // ws layout: [0, 32MiB) W_bf16[N][K]; [32MiB, 96MiB) X_bf16[M][K]
  bf16_t* Wb = (bf16_t*)d_ws;
  bf16_t* Xb = (bf16_t*)((char*)d_ws + (size_t)N_DIM * K_DIM * sizeof(bf16_t));

  dequant_w<<<(N_DIM * (K_DIM / 8)) / 256, 256, 0, stream>>>(wp, cent, scales, Wb);
  x_to_bf16<<<((size_t)M_DIM * K_DIM / 8) / 256, 256, 0, stream>>>(x, Xb);

  dim3 grid(N_DIM / TN, M_DIM / TM);  // 32 x 64 workgroups
  gemm_bf16<<<grid, 256, 0, stream>>>(Xb, Wb, out);
}
